// ConditionalRandomField_55327768708107
// MI455X (gfx1250) — matrix operand order repaired and checked by reading, not compiled, not run on hardware
//
#include <hip/hip_runtime.h>
#include <hip/hip_fp16.h>

typedef __attribute__((ext_vector_type(16))) _Float16 v16h;
typedef __attribute__((ext_vector_type(8)))  _Float16 v8h;
typedef __attribute__((ext_vector_type(8)))  float    v8f;

#define S_LEN 1024
#define T_DIM 64
#define B_TOT 512
#define ROWS  16          // batches per workgroup (WMMA M)
#define P_STR 72          // p row stride in halves (144B: 16B-aligned, bank-skewed)
#define A_STR 68          // alpha row stride in floats (272B: 16B-aligned)

union AV  { v16h v; v8h h[2]; };
union F8  { float4 v[2]; float f[8]; };

#if __has_builtin(__builtin_amdgcn_logf)
__device__ __forceinline__ float fast_ln(float x) {
  return __builtin_amdgcn_logf(x) * 0.69314718055994530942f;  // v_log_f32 * ln2
}
#else
__device__ __forceinline__ float fast_ln(float x) { return __logf(x); }
#endif

__global__ __launch_bounds__(128) void crf_fwd_kernel(
    const float* __restrict__ inputs,        // [B,S,T] f32
    const float* __restrict__ trans,         // [T,T]   f32
    const float* __restrict__ startt,        // [T]
    const float* __restrict__ endt,          // [T]
    const int* __restrict__ tags,            // [B,S]
    const unsigned char* __restrict__ mask,  // [B,S] (bool, 1 byte)
    float* __restrict__ ws_partial)          // [gridDim.x]
{
  __shared__ __align__(16) float     alpha[ROWS][A_STR];
  __shared__ __align__(16) _Float16  p[ROWS][P_STR];
  __shared__ __align__(16) float     mrow[ROWS];
  __shared__ __align__(8)  unsigned char mbyte[ROWS];
  __shared__ float                   logden[ROWS];
  __shared__ float                   red[ROWS][8];
  __shared__ float                   redc[ROWS][8];
  __shared__ float                   contrib[ROWS];

  const int tid  = threadIdx.x;
  const int lane = tid & 31;
  const int wv   = tid >> 5;        // N-tile: states [wv*16, wv*16+16)
  const int bg   = blockIdx.x;      // batch group: batches [bg*16, bg*16+16)
  const int half = lane >> 4;       // 0/1
  const int l16  = lane & 15;
  const int col  = wv * 16 + l16;   // this lane's output state column

  // ---- E = exp(transitions), per-wave B tiles resident in VGPRs ----
  // B layout (f16 32x16): lane => col N=l16, halves t => K = half*16 + t
  const int kb = half * 16;
  v16h bm0, bm1;                     // K = 0..31 and K = 32..63
#pragma unroll
  for (int t = 0; t < 16; ++t) {
    bm0[t] = (_Float16)__expf(trans[(kb + t) * T_DIM + col]);
    bm1[t] = (_Float16)__expf(trans[(kb + t + 32) * T_DIM + col]);
  }

  // per-lane emit base pointers for the C-tile rows this lane owns
  const float* embase[8];
#pragma unroll
  for (int r = 0; r < 8; ++r) {
    const int row = r + 8 * half;
    embase[r] = inputs + ((bg * ROWS + row) * S_LEN) * T_DIM + col;
  }
  const unsigned char* mkrow =
      (tid < ROWS) ? (mask + (bg * ROWS + tid) * S_LEN) : mask;

  // ---- alpha0 = start + emit[s=0]; owner lanes keep their cells in regs ----
  float aprev[8];
  {
    const float st = startt[col];
#pragma unroll
    for (int r = 0; r < 8; ++r) {
      aprev[r] = st + embase[r][0];
      alpha[r + 8 * half][col] = aprev[r];
    }
  }
  __syncthreads();

  // ================= forward recursion =================
  for (int s = 1; s < S_LEN; ++s) {
    // issue emit loads early; waits land where values are consumed
    float emitv[8];
#pragma unroll
    for (int r = 0; r < 8; ++r) emitv[r] = embase[r][s * T_DIM];
    if (tid < ROWS) mbyte[tid] = mkrow[s];

    // row max, wave-local: wave wv owns rows 4wv..4wv+3; 8 lanes per row
    {
      const int row = wv * 4 + (lane >> 3);
      const int sg  = lane & 7;
      const float4* ap = reinterpret_cast<const float4*>(&alpha[row][sg * 8]);
      float4 a0 = ap[0], a1 = ap[1];
      float mx = fmaxf(fmaxf(fmaxf(a0.x, a0.y), fmaxf(a0.z, a0.w)),
                       fmaxf(fmaxf(a1.x, a1.y), fmaxf(a1.z, a1.w)));
      mx = fmaxf(mx, __shfl_xor(mx, 1));
      mx = fmaxf(mx, __shfl_xor(mx, 2));
      mx = fmaxf(mx, __shfl_xor(mx, 4));
      if (sg == 0) mrow[row] = mx;
    }
    __syncthreads();

    // p = exp(alpha - m) -> f16, one b128 store per thread
    {
      const int row = tid >> 3;
      const int k0  = (tid & 7) * 8;
      const float m = mrow[row];
      const float4* ap = reinterpret_cast<const float4*>(&alpha[row][k0]);
      F8 av; av.v[0] = ap[0]; av.v[1] = ap[1];
      v8h hv;
#pragma unroll
      for (int j = 0; j < 8; ++j) hv[j] = (_Float16)__expf(av.f[j] - m);
      *reinterpret_cast<v8h*>(&p[row][k0]) = hv;
    }
    // owner-lane row maxes (8 contiguous) + packed mask bits, before barrier2
    F8 mreg;
    {
      const float4* mrp = reinterpret_cast<const float4*>(&mrow[8 * half]);
      mreg.v[0] = mrp[0]; mreg.v[1] = mrp[1];
    }
    const unsigned long long mb =
        *reinterpret_cast<const unsigned long long*>(&mbyte[8 * half]);
    __syncthreads();

    // A layout (f16 16x32): lane row M=l16; halves 0..15 => K=half*16+0..15
    // (two contiguous b128 LDS loads each)
    AV a0, a1;
    const _Float16* prow = &p[l16][0];
    a0.h[0] = *(const v8h*)(prow + half * 16);
    a0.h[1] = *(const v8h*)(prow + 8 + half * 16);
    a1.h[0] = *(const v8h*)(prow + 32 + half * 16);
    a1.h[1] = *(const v8h*)(prow + 40 + half * 16);

    v8f c = {};
    c = __builtin_amdgcn_wmma_f32_16x16x32_f16(false, a0.v, false, bm0,
                                               (short)0, c, false, false);
    c = __builtin_amdgcn_wmma_f32_16x16x32_f16(false, a1.v, false, bm1,
                                               (short)0, c, false, false);

    // alpha_new = m + log(q) + emit; branchless masked update, store from regs
#pragma unroll
    for (int r = 0; r < 8; ++r) {
      const float anew = mreg.f[r] + fast_ln(c[r]) + emitv[r];
      const bool on = (mb >> (8 * r)) & 1ull;
      aprev[r] = on ? anew : aprev[r];
      alpha[r + 8 * half][col] = aprev[r];
    }
    __syncthreads();
  }

  // ---- log denominator: logsumexp(alpha + end) per batch row ----
  if (tid < ROWS) {
    float mx = -__builtin_inff();
    for (int j = 0; j < T_DIM; ++j) mx = fmaxf(mx, alpha[tid][j] + endt[j]);
    float sm = 0.f;
    for (int j = 0; j < T_DIM; ++j) sm += __expf(alpha[tid][j] + endt[j] - mx);
    logden[tid] = mx + fast_ln(sm);
  }

  // ---- log numerator: 8 threads per batch, 128 timesteps each ----
  {
    const int row = tid >> 3;
    const int q   = tid & 7;
    const int bb  = bg * ROWS + row;
    const int* tg = tags + bb * S_LEN;
    const unsigned char* mk = mask + bb * S_LEN;
    const float* inb = inputs + bb * S_LEN * T_DIM;
    const int s0 = q * (S_LEN / 8), s1 = s0 + (S_LEN / 8);
    float acc = 0.f;
    int cnt = 0;
    int prev = (s0 > 0) ? tg[s0 - 1] : 0;
    for (int s = s0; s < s1; ++s) {
      const int cur = tg[s];
      const float mf = (float)mk[s];
      cnt += (int)mk[s];
      if (s >= 1) acc += trans[prev * T_DIM + cur] * mf;          // trans * mf[1:]
      if (s <= S_LEN - 2) acc += inb[s * T_DIM + cur] * mf;       // emit  * mf[:-1]
      prev = cur;
    }
    red[row][q]  = acc;
    redc[row][q] = (float)cnt;
  }
  __syncthreads();

  if (tid < ROWS) {
    const int bb = bg * ROWS + tid;
    float score = startt[tags[bb * S_LEN]];
    int cnt = 0;
    for (int q = 0; q < 8; ++q) { score += red[tid][q]; cnt += (int)redc[tid][q]; }
    const int last_idx = cnt - 1;
    const int last_tag = tags[bb * S_LEN + last_idx];
    const float mfl = (float)mask[bb * S_LEN + (S_LEN - 1)];
    score += endt[last_tag] +
             inputs[(bb * S_LEN + (S_LEN - 1)) * T_DIM + last_tag] * mfl;
    contrib[tid] = score - logden[tid];
  }
  __syncthreads();

  if (tid == 0) {
    float sm = 0.f;
    for (int r = 0; r < ROWS; ++r) sm += contrib[r];
    ws_partial[bg] = sm;
  }
}

__global__ void crf_reduce_kernel(const float* __restrict__ ws,
                                  float* __restrict__ out, int n) {
  if (threadIdx.x == 0 && blockIdx.x == 0) {
    float s = 0.f;
    for (int i = 0; i < n; ++i) s += ws[i];   // fixed order -> deterministic
    out[0] = s;
  }
}

extern "C" void kernel_launch(void* const* d_in, const int* in_sizes, int n_in,
                              void* d_out, int out_size, void* d_ws, size_t ws_size,
                              hipStream_t stream) {
  (void)in_sizes; (void)n_in; (void)out_size; (void)ws_size;
  const float* inputs         = (const float*)d_in[0];
  const float* trans          = (const float*)d_in[1];
  const float* startt         = (const float*)d_in[2];
  const float* endt           = (const float*)d_in[3];
  const int* tags             = (const int*)d_in[4];
  const unsigned char* maskp  = (const unsigned char*)d_in[5];
  float* ws = (float*)d_ws;

  const int ngroups = B_TOT / ROWS;  // 32 workgroups
  crf_fwd_kernel<<<ngroups, 128, 0, stream>>>(inputs, trans, startt, endt,
                                              tags, maskp, ws);
  crf_reduce_kernel<<<1, 32, 0, stream>>>(ws, (float*)d_out, ngroups);
}
